// Decoder_21088289423384
// MI455X (gfx1250) — matrix operand order repaired and checked by reading, not compiled, not run on hardware
//
#include <hip/hip_runtime.h>

// ---------------------------------------------------------------------------
// Types for CDNA5 WMMA (wave32): A/B = v16h (8 VGPRs), C/D = v8f (8 VGPRs)
// ---------------------------------------------------------------------------
typedef _Float16 half_t;
typedef __attribute__((ext_vector_type(16))) _Float16 v16h;
typedef __attribute__((ext_vector_type(8)))  _Float16 v8h;
typedef __attribute__((ext_vector_type(8)))  float    v8f;

#define GF_RELU 1
#define GF_HALF 2

#define WMMA_F16(a, b, c) \
  __builtin_amdgcn_wmma_f32_16x16x32_f16(false, (a), false, (b), (short)0, (c), false, false)

// ---------------------------------------------------------------------------
// Fragment loads in contiguous k order (lane half h holds k = 16h..16h+15, A and B alike) (see round-1 notes).
// Per-lane base pointers are hoisted; kb is the only loop-varying term.
// ---------------------------------------------------------------------------
__device__ __forceinline__ v16h frag_a(const half_t* __restrict__ p, int kb) {
  v8h lo = *(const v8h*)(p + kb);
  v8h hi = *(const v8h*)(p + kb + 8);
  v16h out;
#pragma unroll
  for (int i = 0; i < 8; ++i) { out[i] = lo[i]; out[8 + i] = hi[i]; }
  return out;
}

__device__ __forceinline__ v16h frag_b(const half_t* __restrict__ p, int kb) {
  return *(const v16h*)(p + kb);
}

// ---------------------------------------------------------------------------
// Generic NT GEMM: C[M,N] = A[MxK](f16) * W[NxK](f16)^T + bias, opt ReLU,
// f32 or f16 store, batched via blockIdx.z with element strides sA/sB/sC.
// Block = 256 threads = 8 waves (2 M x 4 N). Wave tile = 32x64 (8 WMMA/K-step).
// Block tile = 64(M) x 256(N). Requires N % 256 == 0, K % 32 == 0, K >= 32.
// Software-pipelined: next K-slice fragments load while current WMMAs issue.
// ---------------------------------------------------------------------------
__global__ __launch_bounds__(256) void gemm_nt_kernel(
    const half_t* __restrict__ A, const half_t* __restrict__ Bw,
    const float* __restrict__ bias, void* __restrict__ Cv,
    int M, int N, int K, int lda, int ldb, int ldc,
    long long sA, long long sB, long long sC, int flags)
{
  const int z = blockIdx.z;
  A  += (long long)z * sA;
  Bw += (long long)z * sB;

  const int lane = threadIdx.x & 31;
  const int wave = threadIdx.x >> 5;   // 0..7
  const int wm   = wave >> 2;          // 0..1
  const int wn   = wave & 3;           // 0..3
  const int row0 = blockIdx.y * 64 + wm * 32;
  const int col0 = blockIdx.x * 256 + wn * 64;

  const int ml  = lane & 15;           // A: M within tile / B,C: N within tile
  const int khi = lane >> 4;

  // Per-lane base pointers (row-clamped so OOB M reads stay in-buffer).
  int r0 = row0 + ml;      if (r0 > M - 1) r0 = M - 1;
  int r1 = row0 + 16 + ml; if (r1 > M - 1) r1 = M - 1;
  const half_t* pa0 = A + (long long)r0 * lda + khi * 16;
  const half_t* pa1 = A + (long long)r1 * lda + khi * 16;
  const half_t* pb0 = Bw + (long long)(col0 +  0 + ml) * ldb + khi * 16;
  const half_t* pb1 = Bw + (long long)(col0 + 16 + ml) * ldb + khi * 16;
  const half_t* pb2 = Bw + (long long)(col0 + 32 + ml) * ldb + khi * 16;
  const half_t* pb3 = Bw + (long long)(col0 + 48 + ml) * ldb + khi * 16;

  v8f c00 = {}, c01 = {}, c02 = {}, c03 = {};
  v8f c10 = {}, c11 = {}, c12 = {}, c13 = {};

  // ---- pipelined K loop: preload slice 0, then load kb while computing kb-32
  v16h a0 = frag_a(pa0, 0), a1 = frag_a(pa1, 0);
  v16h b0 = frag_b(pb0, 0), b1 = frag_b(pb1, 0);
  v16h b2 = frag_b(pb2, 0), b3 = frag_b(pb3, 0);

  for (int kb = 32; kb < K; kb += 32) {
    v16h A0 = frag_a(pa0, kb), A1 = frag_a(pa1, kb);
    v16h B0 = frag_b(pb0, kb), B1 = frag_b(pb1, kb);
    v16h B2 = frag_b(pb2, kb), B3 = frag_b(pb3, kb);
    c00 = WMMA_F16(a0, b0, c00);
    c01 = WMMA_F16(a0, b1, c01);
    c02 = WMMA_F16(a0, b2, c02);
    c03 = WMMA_F16(a0, b3, c03);
    c10 = WMMA_F16(a1, b0, c10);
    c11 = WMMA_F16(a1, b1, c11);
    c12 = WMMA_F16(a1, b2, c12);
    c13 = WMMA_F16(a1, b3, c13);
    a0 = A0; a1 = A1; b0 = B0; b1 = B1; b2 = B2; b3 = B3;
  }
  c00 = WMMA_F16(a0, b0, c00);
  c01 = WMMA_F16(a0, b1, c01);
  c02 = WMMA_F16(a0, b2, c02);
  c03 = WMMA_F16(a0, b3, c03);
  c10 = WMMA_F16(a1, b0, c10);
  c11 = WMMA_F16(a1, b1, c11);
  c12 = WMMA_F16(a1, b2, c12);
  c13 = WMMA_F16(a1, b3, c13);

  // ---- epilogue: element r of v8f at lane l -> M = r + 8*(l>>4), N = l&15
  const int m_off = khi * 8;
  v8f accs[2][4] = {{c00, c01, c02, c03}, {c10, c11, c12, c13}};
#pragma unroll
  for (int mi = 0; mi < 2; ++mi) {
#pragma unroll
    for (int j = 0; j < 4; ++j) {
      const int col = col0 + j * 16 + ml;
      const float bv = bias ? bias[col] : 0.0f;
#pragma unroll
      for (int r = 0; r < 8; ++r) {
        const int rr = row0 + mi * 16 + m_off + r;
        if (rr < M) {
          float v = accs[mi][j][r] + bv;
          if (flags & GF_RELU) v = fmaxf(v, 0.0f);
          const long long idx = (long long)z * sC + (long long)rr * ldc + col;
          if (flags & GF_HALF) ((half_t*)Cv)[idx] = (half_t)v;
          else                 ((float*)Cv)[idx]  = v;
        }
      }
    }
  }
}

// ---------------------------------------------------------------------------
// Support kernels
// ---------------------------------------------------------------------------
__global__ void f32_to_f16_kernel(const float* __restrict__ src,
                                  half_t* __restrict__ dst, long long n) {
  long long i = (long long)blockIdx.x * blockDim.x + threadIdx.x;
  if (i < n) dst[i] = (half_t)src[i];
}

// rows x sk f32 -> rows x dk f16, zero-padded columns
__global__ void pad_cvt_kernel(const float* __restrict__ src,
                               half_t* __restrict__ dst,
                               int rows, int sk, int dk) {
  long long i = (long long)blockIdx.x * blockDim.x + threadIdx.x;
  if (i >= (long long)rows * dk) return;
  int r = (int)(i / dk), c = (int)(i % dk);
  dst[i] = (c < sk) ? (half_t)src[(long long)r * sk + c] : (half_t)0.0f;
}

// x[b*T+t, :Epad] = relu(embed[words[b*T1+t], :E]), zero-padded to Epad
__global__ void embed_relu_kernel(const int* __restrict__ words,
                                  const float* __restrict__ embed,
                                  half_t* __restrict__ x,
                                  int T, int T1, int E, int Epad, long long total) {
  long long i = (long long)blockIdx.x * blockDim.x + threadIdx.x;
  if (i >= total) return;
  int c = (int)(i % Epad);
  long long row = i / Epad;
  int b = (int)(row / T), t = (int)(row % T);
  int tok = words[(long long)b * T1 + t];
  float v = (c < E) ? embed[(long long)tok * E + c] : 0.0f;
  x[i] = (half_t)fmaxf(v, 0.0f);
}

// enc (B,S,D) f32 -> encF (B,S,D) f16 and encT (B,D,S) f16
__global__ void enc_cvt_transpose_kernel(const float* __restrict__ enc,
                                         half_t* __restrict__ encF,
                                         half_t* __restrict__ encT,
                                         int S, int D, long long total) {
  long long i = (long long)blockIdx.x * blockDim.x + threadIdx.x;
  if (i >= total) return;
  int d = (int)(i % D);
  long long bs = i / D;
  int s = (int)(bs % S), b = (int)(bs / S);
  float v = enc[i];
  encF[i] = (half_t)v;
  encT[((long long)b * D + d) * S + s] = (half_t)v;
}

// h0[layer] = eh[0,layer] + eh[1,layer] ; eh is (2,L,B,H)
__global__ void h0_init_kernel(const float* __restrict__ eh,
                               half_t* __restrict__ h,
                               int layer, int Bsz, int H, int L) {
  int i = blockIdx.x * blockDim.x + threadIdx.x;
  if (i >= Bsz * H) return;
  long long n = (long long)Bsz * H;
  float v = eh[(long long)layer * n + i] + eh[(long long)(L + layer) * n + i];
  h[i] = (half_t)v;
}

// GRU gate elementwise; gi rows are (b*T + t), gh rows are b; h is (B,H) f16
__global__ void gru_gates_kernel(const float* __restrict__ gi,
                                 const float* __restrict__ gh,
                                 half_t* __restrict__ h,
                                 half_t* __restrict__ ys,
                                 int t, int T, int Bsz, int H, int ys_ld) {
  int i = blockIdx.x * blockDim.x + threadIdx.x;
  if (i >= Bsz * H) return;
  int b = i / H, j = i % H;
  const float* gib = gi + (long long)(b * T + t) * (3 * H);
  const float* ghb = gh + (long long)b * (3 * H);
  float r  = 1.0f / (1.0f + __expf(-(gib[j]         + ghb[j])));
  float zg = 1.0f / (1.0f + __expf(-(gib[H + j]     + ghb[H + j])));
  float nn = tanhf(gib[2 * H + j] + r * ghb[2 * H + j]);
  float hp = (float)h[i];
  float hn = (1.0f - zg) * nn + zg * hp;
  h[i] = (half_t)hn;
  ys[(long long)(b * T + t) * ys_ld + j] = (half_t)hn;
}

// row softmax over S=256 (blockDim == S), f32 in -> f16 out
__global__ __launch_bounds__(256) void softmax_kernel(const float* __restrict__ scores,
                                                      half_t* __restrict__ attn, int S) {
  const int row = blockIdx.x;
  const int tid = threadIdx.x;
  __shared__ float red[256];
  float v = (tid < S) ? scores[(long long)row * S + tid] : -3.0e38f;
  red[tid] = v; __syncthreads();
  for (int s = 128; s > 0; s >>= 1) { if (tid < s) red[tid] = fmaxf(red[tid], red[tid + s]); __syncthreads(); }
  float mx = red[0]; __syncthreads();
  float e = (tid < S) ? __expf(v - mx) : 0.0f;
  red[tid] = e; __syncthreads();
  for (int s = 128; s > 0; s >>= 1) { if (tid < s) red[tid] += red[tid + s]; __syncthreads(); }
  float inv = 1.0f / red[0];
  if (tid < S) attn[(long long)row * S + tid] = (half_t)(e * inv);
}

// ---------------------------------------------------------------------------
// Host orchestration
// ---------------------------------------------------------------------------
extern "C" void kernel_launch(void* const* d_in, const int* in_sizes, int n_in,
                              void* d_out, int out_size, void* d_ws, size_t ws_size,
                              hipStream_t stream) {
  const int Bsz = 32, S = 256, T = 64, T1 = 65, H = 512, E = 300, Epad = 320;
  const int V = 32000, G3 = 3 * H /*1536*/, D2 = 2 * H /*1024*/, L = 2;
  const int MT = Bsz * T;      // 2048
  const int MS = Bsz * S;      // 8192

  const float* enc_out = (const float*)d_in[0];
  const float* enc_hid = (const float*)d_in[1];
  const int*   words   = (const int*)  d_in[2];
  const float* embed   = (const float*)d_in[3];
  const float* Wih0    = (const float*)d_in[4];
  const float* Whh0    = (const float*)d_in[5];
  const float* bih0    = (const float*)d_in[6];
  const float* bhh0    = (const float*)d_in[7];
  const float* Wih1    = (const float*)d_in[8];
  const float* Whh1    = (const float*)d_in[9];
  const float* bih1    = (const float*)d_in[10];
  const float* bhh1    = (const float*)d_in[11];
  const float* attn_W  = (const float*)d_in[12];
  const float* attn_b  = (const float*)d_in[13];
  const float* aw_W    = (const float*)d_in[14];
  const float* aw_b    = (const float*)d_in[15];
  const float* out_W   = (const float*)d_in[16];
  const float* out_b   = (const float*)d_in[17];
  float* logits = (float*)d_out;

  // ---- workspace carve-out (element counts; 256B aligned) ----
  char* ws = (char*)d_ws;
  size_t cur = 0;
  auto carve = [&](long long elems, int esz) -> void* {
    void* p = ws + cur;
    cur = (cur + (size_t)elems * esz + 255) & ~(size_t)255;
    return p;
  };
  half_t* wih0h = (half_t*)carve((long long)G3 * Epad, 2);
  half_t* whh0h = (half_t*)carve((long long)G3 * H, 2);
  half_t* wih1h = (half_t*)carve((long long)G3 * H, 2);
  half_t* whh1h = (half_t*)carve((long long)G3 * H, 2);
  half_t* attnWh= (half_t*)carve((long long)H * D2, 2);
  half_t* awWh  = (half_t*)carve((long long)G3 * G3, 2);
  half_t* outWh = (half_t*)carve((long long)V * G3, 2);
  half_t* encFh = (half_t*)carve((long long)MS * D2, 2);
  half_t* encTh = (half_t*)carve((long long)Bsz * D2 * S, 2);
  half_t* xh    = (half_t*)carve((long long)MT * Epad, 2);
  float*  gi    = (float*) carve((long long)MT * G3, 4);
  float*  gh    = (float*) carve((long long)Bsz * G3, 4);
  half_t* hbuf  = (half_t*)carve((long long)Bsz * H, 2);
  half_t* ys0h  = (half_t*)carve((long long)MT * H, 2);
  half_t* cath  = (half_t*)carve((long long)MT * G3, 2);
  half_t* projh = (half_t*)carve((long long)MS * H, 2);
  float*  scor  = (float*) carve((long long)MT * S, 4);
  half_t* attnh = (half_t*)carve((long long)MT * S, 2);
  half_t* oh    = (half_t*)carve((long long)MT * G3, 2);

  auto cvt = [&](const float* s, half_t* d, long long n) {
    f32_to_f16_kernel<<<(unsigned)((n + 255) / 256), 256, 0, stream>>>(s, d, n);
  };
  auto gemm = [&](const half_t* A, const half_t* Bw, const float* bias, void* C,
                  int M, int N, int K, int lda, int ldb, int ldc,
                  long long sA, long long sB, long long sC, int Z, int flags) {
    dim3 g((unsigned)(N / 256), (unsigned)((M + 63) / 64), (unsigned)Z);
    gemm_nt_kernel<<<g, 256, 0, stream>>>(A, Bw, bias, C, M, N, K,
                                          lda, ldb, ldc, sA, sB, sC, flags);
  };

  // ---- weight & input precision conversion (once per call) ----
  pad_cvt_kernel<<<(G3 * Epad + 255) / 256, 256, 0, stream>>>(Wih0, wih0h, G3, E, Epad);
  cvt(Whh0, whh0h, (long long)G3 * H);
  cvt(Wih1, wih1h, (long long)G3 * H);
  cvt(Whh1, whh1h, (long long)G3 * H);
  cvt(attn_W, attnWh, (long long)H * D2);
  cvt(aw_W, awWh, (long long)G3 * G3);
  cvt(out_W, outWh, (long long)V * G3);
  {
    long long tot = (long long)MS * D2;
    enc_cvt_transpose_kernel<<<(unsigned)((tot + 255) / 256), 256, 0, stream>>>(
        enc_out, encFh, encTh, S, D2, tot);
  }
  {
    long long tot = (long long)MT * Epad;
    embed_relu_kernel<<<(unsigned)((tot + 255) / 256), 256, 0, stream>>>(
        words, embed, xh, T, T1, E, Epad, tot);
  }

  // ---- GRU layer 0 ----
  gemm(xh, wih0h, bih0, gi, MT, G3, Epad, Epad, Epad, G3, 0, 0, 0, 1, 0);
  h0_init_kernel<<<(Bsz * H + 255) / 256, 256, 0, stream>>>(enc_hid, hbuf, 0, Bsz, H, L);
  for (int t = 0; t < T; ++t) {
    gemm(hbuf, whh0h, bhh0, gh, Bsz, G3, H, H, H, G3, 0, 0, 0, 1, 0);
    gru_gates_kernel<<<(Bsz * H + 255) / 256, 256, 0, stream>>>(
        gi, gh, hbuf, ys0h, t, T, Bsz, H, H);
  }

  // ---- GRU layer 1 (hidden states written into cat[:, 0:512], ld = 1536) ----
  gemm(ys0h, wih1h, bih1, gi, MT, G3, H, H, H, G3, 0, 0, 0, 1, 0);
  h0_init_kernel<<<(Bsz * H + 255) / 256, 256, 0, stream>>>(enc_hid, hbuf, 1, Bsz, H, L);
  for (int t = 0; t < T; ++t) {
    gemm(hbuf, whh1h, bhh1, gh, Bsz, G3, H, H, H, G3, 0, 0, 0, 1, 0);
    gru_gates_kernel<<<(Bsz * H + 255) / 256, 256, 0, stream>>>(
        gi, gh, hbuf, cath, t, T, Bsz, H, G3);
  }

  // ---- attention ----
  // proj[b,s,:] = enc[b,s,:] @ attn_W^T + attn_b   (f16 out)
  gemm(encFh, attnWh, attn_b, projh, MS, H, D2, D2, D2, H, 0, 0, 0, 1, GF_HALF);
  // scores[b] = h[b] (64x512, lda=1536 inside cat) @ proj[b]^T  (f32)
  gemm(cath, projh, nullptr, scor, T, S, H, G3, H, S,
       (long long)T * G3, (long long)S * H, (long long)T * S, Bsz, 0);
  softmax_kernel<<<MT, 256, 0, stream>>>(scor, attnh, S);
  // ctx[b] = attn[b] (64x256) @ encT[b]^T (1024x256) -> cat[:, 512:1536] (f16)
  gemm(attnh, encTh, nullptr, cath + H, T, D2, S, S, S, G3,
       (long long)T * S, (long long)D2 * S, (long long)T * G3, Bsz, GF_HALF);

  // ---- output head ----
  // o = relu(cat @ aw_W^T + aw_b)  (f16)
  gemm(cath, awWh, aw_b, oh, MT, G3, G3, G3, G3, G3, 0, 0, 0, 1, GF_RELU | GF_HALF);
  // logits = o @ out_W^T + out_b   (f32, straight to d_out)
  gemm(oh, outWh, out_b, logits, MT, V, G3, G3, G3, V, 0, 0, 0, 1, 0);
}
